// DLASSO_GNNHyp3_Progressive_10677288698529
// MI455X (gfx1250) — matrix operand order repaired and checked by reading, not compiled, not run on hardware
//
#include <hip/hip_runtime.h>
#include <math.h>

#define B_DIM 64
#define P_DIM 50
#define M_DIM 100
#define N_DIM 200
#define HF    128
#define KIT   5
#define E_DIM 400
#define F5    512              // 4*H
#define BN_SCALE 0.9999950000374997f   // 1/sqrt(1+1e-5)

typedef __attribute__((ext_vector_type(2)))  float          v2f;
typedef __attribute__((ext_vector_type(4)))  float          v4f;
typedef __attribute__((ext_vector_type(8)))  float          v8f;
typedef __attribute__((ext_vector_type(16))) __bf16         v16bf;
typedef __attribute__((ext_vector_type(16))) unsigned short v16us;

// ---------- helpers ----------
__device__ __forceinline__ unsigned short f2bf_u(float f) {
  union { float f; unsigned u; } v; v.f = f;
  unsigned r = v.u + 0x7FFFu + ((v.u >> 16) & 1u);   // round-to-nearest-even
  return (unsigned short)(r >> 16);
}
__device__ __forceinline__ v16bf us_to_bf(v16us u) {
  union { v16us u; v16bf b; } c; c.u = u; return c.b;
}
__device__ __forceinline__ float lrelu(float v) { return (v >= 0.f) ? v : 0.01f * v; }
__device__ __forceinline__ float clipf(float v, float lo, float hi) { return fminf(fmaxf(v, lo), hi); }
__device__ __forceinline__ int imin(int a, int b) { return a < b ? a : b; }

// ---------- small utility kernels ----------
__global__ void k_copy(const float* __restrict__ s, float* __restrict__ d, int n) {
  int i = blockIdx.x * blockDim.x + threadIdx.x;
  if (i < n) d[i] = s[i];
}
__global__ void k_zero(float* __restrict__ d, int n) {
  int i = blockIdx.x * blockDim.x + threadIdx.x;
  if (i < n) d[i] = 0.f;
}
// transpose + cast weights to bf16: dst[n*KdPad + k] = bf16(src[k*Nc + n]); zero K-pad
__global__ void k_cast_bf16_t(const float* __restrict__ src, unsigned short* __restrict__ dst,
                              int Kd, int Nc, int KdPad) {
  long long idx = (long long)blockIdx.x * blockDim.x + threadIdx.x;
  long long total = (long long)Nc * KdPad;
  if (idx >= total) return;
  int n = (int)(idx / KdPad), k = (int)(idx % KdPad);
  dst[idx] = (k < Kd) ? f2bf_u(src[(size_t)k * Nc + n]) : (unsigned short)0;
}

// ---------- per-batch graph statistics: deg (via d), nrm, sum_nb (via s) ----------
__global__ void k_graph_stats(const int* __restrict__ edge, float* __restrict__ invdeg,
                              float* __restrict__ nrm, float* __restrict__ sumnb) {
  __shared__ float degs[P_DIM];
  __shared__ float snb[P_DIM];
  int bb = blockIdx.x;
  const int* sArr = edge + (size_t)bb * 2 * E_DIM;
  const int* dArr = sArr + E_DIM;
  for (int i = threadIdx.x; i < P_DIM; i += blockDim.x) { degs[i] = 0.f; snb[i] = 0.f; }
  __syncthreads();
  for (int e = threadIdx.x; e < E_DIM; e += blockDim.x) {
    atomicAdd(&degs[dArr[e]], 1.f);
    atomicAdd(&snb[sArr[e]], 1.f);
  }
  __syncthreads();
  for (int i = threadIdx.x; i < P_DIM; i += blockDim.x) {
    float dg = degs[i] + 1.f;
    degs[i] = dg;
    invdeg[(size_t)bb * P_DIM + i] = 1.f / dg;
    sumnb [(size_t)bb * P_DIM + i] = snb[i];
  }
  __syncthreads();
  for (int e = threadIdx.x; e < E_DIM; e += blockDim.x)
    nrm[(size_t)bb * E_DIM + e] = rsqrtf(degs[sArr[e]] * degs[dArr[e]]);
}

// ---------- AtA[p] = A_p^T A_p  (fp32 WMMA 16x16x4, branch-free inner loop) ----------
__global__ void k_ata(const float* __restrict__ A0, float* __restrict__ AtA) {
  const int NT = 13;                       // ceil(200/16)
  int wave = threadIdx.x >> 5, lane = threadIdx.x & 31;
  int tile = blockIdx.x * 8 + wave;
  if (tile >= NT * NT) return;
  int p = blockIdx.y;
  int ti = tile / NT, tj = tile % NT;
  int lo = lane & 15, hi = lane >> 4;
  const float* Ap = A0 + (size_t)p * M_DIM * N_DIM;
  int ig = ti * 16 + lo, jg = tj * 16 + lo;       // for store guards
  int i = imin(ig, N_DIM - 1);                    // clamped (garbage rows never stored)
  int j = imin(jg, N_DIM - 1);
  v8f c = {};
  for (int k0 = 0; k0 < M_DIM; k0 += 4) {         // 100 % 4 == 0
    int m0 = k0 + hi * 2;
    v2f a, b;
    a.x = Ap[(size_t)m0 * N_DIM + i];
    a.y = Ap[(size_t)(m0 + 1) * N_DIM + i];
    b.x = Ap[(size_t)m0 * N_DIM + j];
    b.y = Ap[(size_t)(m0 + 1) * N_DIM + j];
    c = __builtin_amdgcn_wmma_f32_16x16x4_f32(false, a, false, b, (short)0, c, false, false);
  }
  float* Cp = AtA + (size_t)p * N_DIM * N_DIM;
  #pragma unroll
  for (int r = 0; r < 8; ++r) {
    int row = ti * 16 + r + 8 * hi;
    if (row < N_DIM && jg < N_DIM) Cp[(size_t)row * N_DIM + jg] = c[r];
  }
}

// ---------- Atb: per p, (64x100) @ (100x200)  (fp32 WMMA, branch-free) ----------
__global__ void k_atb(const float* __restrict__ A0, const float* __restrict__ bin,
                      float* __restrict__ Atb) {
  const int NT = 13, MT = 4;
  int wave = threadIdx.x >> 5, lane = threadIdx.x & 31;
  int tile = blockIdx.x * 8 + wave;
  if (tile >= MT * NT) return;
  int p = blockIdx.y;
  int ti = tile / NT, tj = tile % NT;
  int lo = lane & 15, hi = lane >> 4;
  const float* Ap = A0 + (size_t)p * M_DIM * N_DIM;
  int row = ti * 16 + lo;                        // batch (< 64 always)
  int colg = tj * 16 + lo;
  int col = imin(colg, N_DIM - 1);
  const float* brow = bin + ((size_t)row * P_DIM + p) * M_DIM;
  v8f c = {};
  for (int k0 = 0; k0 < M_DIM; k0 += 4) {
    int m0 = k0 + hi * 2;
    v2f a = *(const v2f*)(brow + m0);            // contiguous pair
    v2f b;
    b.x = Ap[(size_t)m0 * N_DIM + col];
    b.y = Ap[(size_t)(m0 + 1) * N_DIM + col];
    c = __builtin_amdgcn_wmma_f32_16x16x4_f32(false, a, false, b, (short)0, c, false, false);
  }
  #pragma unroll
  for (int r = 0; r < 8; ++r) {
    int rr = ti * 16 + r + 8 * hi;
    if (colg < N_DIM) Atb[((size_t)rr * P_DIM + p) * N_DIM + colg] = c[r];
  }
}

// ---------- AtAy into Xn[...,0:200]: C[b,i]=sum_j y[b,j]*AtA[i,j]  (fp32 WMMA) ----------
__global__ void k_atay(const float* __restrict__ AtA, const float* __restrict__ y,
                       float* __restrict__ Xn) {
  const int NT = 13, MT = 4;
  int wave = threadIdx.x >> 5, lane = threadIdx.x & 31;
  int tile = blockIdx.x * 8 + wave;
  if (tile >= MT * NT) return;
  int p = blockIdx.y;
  int ti = tile / NT, tj = tile % NT;
  int lo = lane & 15, hi = lane >> 4;
  const float* Ap = AtA + (size_t)p * N_DIM * N_DIM;
  int row = ti * 16 + lo;                        // batch
  int colg = tj * 16 + lo;
  int col = imin(colg, N_DIM - 1);
  const float* yrow = y + ((size_t)row * P_DIM + p) * N_DIM;
  const float* bptr = Ap + (size_t)col * N_DIM;  // AtA^T access (row "col")
  v8f c = {};
  for (int k0 = 0; k0 < N_DIM; k0 += 4) {        // 200 % 4 == 0
    int jk = k0 + hi * 2;
    v2f a = *(const v2f*)(yrow + jk);            // contiguous pair
    v2f b = *(const v2f*)(bptr + jk);            // contiguous pair
    c = __builtin_amdgcn_wmma_f32_16x16x4_f32(false, a, false, b, (short)0, c, false, false);
  }
  #pragma unroll
  for (int r = 0; r < 8; ++r) {
    int rr = ti * 16 + r + 8 * hi;
    if (colg < N_DIM) Xn[((size_t)rr * P_DIM + p) * (2 * N_DIM) + colg] = c[r];
  }
}

__global__ void k_atb_to_xn(const float* __restrict__ Atb, float* __restrict__ Xn) {
  int i = blockIdx.x * blockDim.x + threadIdx.x;
  int total = B_DIM * P_DIM * N_DIM;
  if (i >= total) return;
  int n = i % N_DIM; size_t bp = (size_t)i / N_DIM;
  Xn[bp * (2 * N_DIM) + N_DIM + n] = Atb[i];
}

// ---------- bf16 WMMA GEMM: C(Mr x Nc) = X(Mr x Kd, f32) @ W^T(Nc x KdPad, bf16) [+bias]
// Weights transposed + K-padded with zeros => branch-free, fully vectorized inner loop.
__global__ void k_gemm_bf16(const float* __restrict__ X, const unsigned short* __restrict__ Wt,
                            const float* __restrict__ bias, float* __restrict__ C,
                            int Mr, int Nc, int Kd, int KdPad) {
  int wave = threadIdx.x >> 5, lane = threadIdx.x & 31;
  int nt = (Nc + 15) >> 4, mt = (Mr + 15) >> 4;
  int tile = blockIdx.x * 8 + wave;
  if (tile >= mt * nt) return;
  int ti = tile / nt, tj = tile % nt;
  int lo = lane & 15, hi = lane >> 4;
  int arow = ti * 16 + lo;
  int bcol = tj * 16 + lo;
  int arowc = imin(arow, Mr - 1);
  int bcolc = imin(bcol, Nc - 1);
  const float* xrow = X + (size_t)arowc * Kd + hi * 16;
  const unsigned short* wrow = Wt + (size_t)bcolc * KdPad + hi * 16;
  v8f acc = {};
  for (int k0 = 0; k0 < Kd; k0 += 32) {
    // A fragment: one contiguous 16-float run per lane -> 4x b128 loads
    v4f a0 = *(const v4f*)(xrow + k0);
    v4f a1 = *(const v4f*)(xrow + k0 + 4);
    v4f a2 = *(const v4f*)(xrow + k0 + 8);
    v4f a3 = *(const v4f*)(xrow + k0 + 12);
    v16us au;
    #pragma unroll
    for (int e = 0; e < 4; ++e) {
      au[e]      = f2bf_u(a0[e]);
      au[e + 4]  = f2bf_u(a1[e]);
      au[e + 8]  = f2bf_u(a2[e]);
      au[e + 12] = f2bf_u(a3[e]);
    }
    // B fragment: 16 consecutive bf16 per lane (transposed weights)
    v16us bu = *(const v16us*)(wrow + k0);
    acc = __builtin_amdgcn_wmma_f32_16x16x32_bf16(false, us_to_bf(au), false, us_to_bf(bu),
                                                  (short)0, acc, false, false);
  }
  #pragma unroll
  for (int r = 0; r < 8; ++r) {
    int row = ti * 16 + r + 8 * hi;
    if (row < Mr && bcol < Nc) {
      float v = acc[r];
      if (bias) v += bias[bcol];
      C[(size_t)row * Nc + bcol] = v;
    }
  }
}

// ---------- GCN aggregation ----------
__global__ void k_agg_init(const float* __restrict__ H, const float* __restrict__ invdeg,
                           float* __restrict__ AGG, int F) {
  long long idx = (long long)blockIdx.x * blockDim.x + threadIdx.x;
  long long total = (long long)B_DIM * P_DIM * F;
  if (idx < total) AGG[idx] = H[idx] * invdeg[idx / F];
}
__global__ void k_gcn_scatter(const float* __restrict__ H, const int* __restrict__ edge,
                              const float* __restrict__ nrm, float* __restrict__ AGG, int F) {
  int bb = blockIdx.x, e = blockIdx.y;
  const int* sArr = edge + (size_t)bb * 2 * E_DIM;
  int sp = sArr[e], dp = sArr[E_DIM + e];
  float w = nrm[(size_t)bb * E_DIM + e];
  const float* hs = H + ((size_t)bb * P_DIM + sp) * F;
  float* ad = AGG + ((size_t)bb * P_DIM + dp) * F;
  for (int f = threadIdx.x; f < F; f += blockDim.x) atomicAdd(&ad[f], w * hs[f]);
}
__global__ void k_gcn_finish(const float* __restrict__ AGG, const float* __restrict__ bias,
                             const float* __restrict__ g, const float* __restrict__ be,
                             float* __restrict__ Xout, int F) {
  long long idx = (long long)blockIdx.x * blockDim.x + threadIdx.x;
  long long total = (long long)B_DIM * P_DIM * F;
  if (idx >= total) return;
  int f = (int)(idx % F);
  float v = lrelu(AGG[idx] + bias[f]);
  Xout[idx] = v * (g[f] * BN_SCALE) + be[f];
}

// ---------- row LayerNorm (block=256 per row), optional lrelu ----------
__global__ void k_layernorm(const float* __restrict__ Xin, float* __restrict__ Xout,
                            const float* __restrict__ g, const float* __restrict__ bta,
                            int F, int do_lrelu) {
  __shared__ float red[256];
  int row = blockIdx.x, t = threadIdx.x;
  const float* x = Xin + (size_t)row * F;
  float* o = Xout + (size_t)row * F;
  float s = 0.f;
  for (int f = t; f < F; f += 256) s += x[f];
  red[t] = s; __syncthreads();
  for (int st = 128; st > 0; st >>= 1) { if (t < st) red[t] += red[t + st]; __syncthreads(); }
  float mu = red[0] / F; __syncthreads();
  float vs = 0.f;
  for (int f = t; f < F; f += 256) { float d = x[f] - mu; vs += d * d; }
  red[t] = vs; __syncthreads();
  for (int st = 128; st > 0; st >>= 1) { if (t < st) red[t] += red[t + st]; __syncthreads(); }
  float rstd = rsqrtf(red[0] / F + 1e-5f);
  for (int f = t; f < F; f += 256) {
    float v = (x[f] - mu) * rstd * g[f] + bta[f];
    if (do_lrelu) v = lrelu(v);
    o[f] = v;
  }
}

__global__ void k_sig(const float* __restrict__ in, float* __restrict__ out, int n) {
  int i = blockIdx.x * blockDim.x + threadIdx.x;
  if (i < n) {
    float v = 1.f / (1.f + __expf(-in[i]));
    out[i] = clipf(v, 1e-4f, 0.9999f);
  }
}

// ---------- solver update ----------
__global__ void k_grad_y(const float* __restrict__ Xn, const float* __restrict__ Atb,
                         float* __restrict__ y, const float* __restrict__ U,
                         const float* __restrict__ dlt, const float* __restrict__ sumnb,
                         const float* __restrict__ hyp, float* __restrict__ Yout) {
  int idx = blockIdx.x * blockDim.x + threadIdx.x;
  int total = B_DIM * P_DIM * N_DIM;
  if (idx >= total) return;
  int n = idx % N_DIM;
  size_t bp = (size_t)idx / N_DIM;
  int bb = (int)(bp / P_DIM);
  float alpha = hyp[bb * 4 + 0];
  float tau   = fminf(hyp[bb * 4 + 1], 0.9999f);
  float rho   = fminf(hyp[bb * 4 + 2], 0.9999f);
  float atay  = Xn[bp * (2 * N_DIM) + n];
  float yv    = y[idx];
  float sgn   = (float)((yv > 0.f) - (yv < 0.f));
  float grad  = atay - Atb[idx] + sgn * tau + U[idx] * sumnb[bp] + dlt[idx] * rho;
  grad = clipf(grad, -10.f, 10.f);
  float yn = clipf(yv - alpha * grad, -100.f, 100.f);
  y[idx] = yn;
  Yout[idx] = yn;
}
__global__ void k_delta_scatter(const float* __restrict__ y, const int* __restrict__ edge,
                                float* __restrict__ dlt) {
  int bb = blockIdx.x, e = blockIdx.y;
  const int* sArr = edge + (size_t)bb * 2 * E_DIM;
  int sp = sArr[e], dp = sArr[E_DIM + e];
  const float* ys = y + ((size_t)bb * P_DIM + sp) * N_DIM;
  const float* yd = y + ((size_t)bb * P_DIM + dp) * N_DIM;
  float* ds = dlt + ((size_t)bb * P_DIM + sp) * N_DIM;
  float* dd = dlt + ((size_t)bb * P_DIM + dp) * N_DIM;
  for (int n = threadIdx.x; n < N_DIM; n += blockDim.x) {
    float diff = ys[n] - yd[n];
    atomicAdd(&ds[n],  diff);
    atomicAdd(&dd[n], -diff);
  }
}
__global__ void k_u_update(float* __restrict__ dlt, float* __restrict__ U,
                           const float* __restrict__ hyp) {
  int idx = blockIdx.x * blockDim.x + threadIdx.x;
  int total = B_DIM * P_DIM * N_DIM;
  if (idx >= total) return;
  int bb = idx / (P_DIM * N_DIM);
  float eta = fminf(hyp[bb * 4 + 3], 0.9999f);
  float dv = clipf(dlt[idx], -20.f, 20.f);
  dlt[idx] = dv;
  U[idx] = clipf(U[idx] + dv * eta, -100.f, 100.f);
}

// ===================== host =====================
extern "C" void kernel_launch(void* const* d_in, const int* in_sizes, int n_in,
                              void* d_out, int out_size, void* d_ws, size_t ws_size,
                              hipStream_t stream) {
  (void)in_sizes; (void)n_in; (void)out_size; (void)ws_size;
  const float* bin  = (const float*)d_in[0];
  const int*   edge = (const int*)d_in[1];
  const float* A0   = (const float*)d_in[2];
  const float *Wg[5], *bg[5], *gg[5], *beg[5];
  for (int i = 0; i < 5; ++i) {
    Wg[i]  = (const float*)d_in[3 + 4 * i];
    bg[i]  = (const float*)d_in[4 + 4 * i];
    gg[i]  = (const float*)d_in[5 + 4 * i];
    beg[i] = (const float*)d_in[6 + 4 * i];
  }
  const float* ln_g = (const float*)d_in[23];
  const float* ln_b = (const float*)d_in[24];
  const float* Wd1 = (const float*)d_in[25]; const float* bd1 = (const float*)d_in[26];
  const float* l1g = (const float*)d_in[27]; const float* l1b = (const float*)d_in[28];
  const float* Wd2 = (const float*)d_in[29]; const float* bd2 = (const float*)d_in[30];
  const float* l2g = (const float*)d_in[31]; const float* l2b = (const float*)d_in[32];
  const float* Wd3 = (const float*)d_in[33]; const float* bd3 = (const float*)d_in[34];
  const float* l3g = (const float*)d_in[35]; const float* l3b = (const float*)d_in[36];
  const float* Wfc = (const float*)d_in[37]; const float* bfc = (const float*)d_in[38];
  const float* y0  = (const float*)d_in[39];
  const float* U0  = (const float*)d_in[40];
  const float* dl0 = (const float*)d_in[41];
  float* out = (float*)d_out;

  // workspace allocation (256B aligned blocks)
  char* base = (char*)d_ws; size_t off = 0;
  auto alloc = [&](size_t bytes) -> void* {
    void* p = base + off; off += (bytes + 255) & ~(size_t)255; return p;
  };
  const int BPN = B_DIM * P_DIM * N_DIM;           // 640000
  const int BPF = B_DIM * P_DIM * F5;              // 1638400
  float* AtA   = (float*)alloc((size_t)P_DIM * N_DIM * N_DIM * 4);
  float* Atb   = (float*)alloc((size_t)BPN * 4);
  float* yS    = (float*)alloc((size_t)BPN * 4);
  float* US    = (float*)alloc((size_t)BPN * 4);
  float* dltS  = (float*)alloc((size_t)BPN * 4);
  float* invdeg= (float*)alloc((size_t)B_DIM * P_DIM * 4);
  float* nrm   = (float*)alloc((size_t)B_DIM * E_DIM * 4);
  float* sumnb = (float*)alloc((size_t)B_DIM * P_DIM * 4);
  float* Xn    = (float*)alloc((size_t)B_DIM * P_DIM * 2 * N_DIM * 4);
  float* buf0  = (float*)alloc((size_t)BPF * 4);
  float* buf1  = (float*)alloc((size_t)BPF * 4);
  float* bufH  = (float*)alloc((size_t)BPF * 4);
  float* bufA  = (float*)alloc((size_t)BPF * 4);
  float* z1r = (float*)alloc(64 * 512 * 4); float* z1 = (float*)alloc(64 * 512 * 4);
  float* z2r = (float*)alloc(64 * 256 * 4); float* z2 = (float*)alloc(64 * 256 * 4);
  float* z3r = (float*)alloc(64 * 128 * 4); float* z3 = (float*)alloc(64 * 128 * 4);
  float* hyr = (float*)alloc(64 * 4 * 4);   float* hyp = (float*)alloc(64 * 4 * 4);

  // transposed bf16 weights: [Nc][KdPad], KdPad = roundup(Kd,32), zero-padded
  const int wKd[9]  = {400, 128, 256, 512, 512, 25600, 512, 256, 128};
  const int wNc[9]  = {128, 256, 512, 512, 512, 512, 256, 128, 4};
  int wKdP[9];
  const float* wsrc[9] = {Wg[0], Wg[1], Wg[2], Wg[3], Wg[4], Wd1, Wd2, Wd3, Wfc};
  unsigned short* wbf[9];
  for (int i = 0; i < 9; ++i) {
    wKdP[i] = (wKd[i] + 31) & ~31;
    wbf[i] = (unsigned short*)alloc((size_t)wNc[i] * wKdP[i] * 2);
  }

  auto eb = [](long long n) { return (unsigned)((n + 255) / 256); };

  // --- setup ---
  for (int i = 0; i < 9; ++i) {
    long long tot = (long long)wNc[i] * wKdP[i];
    k_cast_bf16_t<<<eb(tot), 256, 0, stream>>>(wsrc[i], wbf[i], wKd[i], wNc[i], wKdP[i]);
  }
  k_copy<<<eb(BPN), 256, 0, stream>>>(y0,  yS,   BPN);
  k_copy<<<eb(BPN), 256, 0, stream>>>(U0,  US,   BPN);
  k_copy<<<eb(BPN), 256, 0, stream>>>(dl0, dltS, BPN);
  k_graph_stats<<<B_DIM, 256, 0, stream>>>(edge, invdeg, nrm, sumnb);
  k_ata<<<dim3((13 * 13 + 7) / 8, P_DIM), 256, 0, stream>>>(A0, AtA);
  k_atb<<<dim3((4 * 13 + 7) / 8,  P_DIM), 256, 0, stream>>>(A0, bin, Atb);
  k_atb_to_xn<<<eb(BPN), 256, 0, stream>>>(Atb, Xn);

  auto gemm = [&](const float* X, int wi, const float* bias, float* C, int Mr) {
    int Nc = wNc[wi], Kd = wKd[wi], KdP = wKdP[wi];
    int tiles = ((Mr + 15) / 16) * ((Nc + 15) / 16);
    k_gemm_bf16<<<(tiles + 7) / 8, 256, 0, stream>>>(X, wbf[wi], bias, C, Mr, Nc, Kd, KdP);
  };

  for (int k = 0; k < KIT; ++k) {
    // AtAy -> Xn[:, :200]
    k_atay<<<dim3((4 * 13 + 7) / 8, P_DIM), 256, 0, stream>>>(AtA, yS, Xn);

    // 5 GCN layers
    const float* Xcur = Xn;
    for (int l = 0; l < 5; ++l) {
      float* Xnext = (l & 1) ? buf1 : buf0;
      int Fo = wNc[l];
      gemm(Xcur, l, nullptr, bufH, B_DIM * P_DIM);
      long long tot = (long long)B_DIM * P_DIM * Fo;
      k_agg_init<<<eb(tot), 256, 0, stream>>>(bufH, invdeg, bufA, Fo);
      k_gcn_scatter<<<dim3(B_DIM, E_DIM), 256, 0, stream>>>(bufH, edge, nrm, bufA, Fo);
      k_gcn_finish<<<eb(tot), 256, 0, stream>>>(bufA, bg[l], gg[l], beg[l], Xnext, Fo);
      Xcur = Xnext;
    }
    // final LayerNorm over features -> enc (B, P*512) contiguous in bufH
    k_layernorm<<<B_DIM * P_DIM, 256, 0, stream>>>(Xcur, bufH, ln_g, ln_b, F5, 0);

    // decoder MLP (bf16 WMMA GEMMs + LN + lrelu)
    gemm(bufH, 5, bd1, z1r, B_DIM);
    k_layernorm<<<B_DIM, 256, 0, stream>>>(z1r, z1, l1g, l1b, 512, 1);
    gemm(z1, 6, bd2, z2r, B_DIM);
    k_layernorm<<<B_DIM, 256, 0, stream>>>(z2r, z2, l2g, l2b, 256, 1);
    gemm(z2, 7, bd3, z3r, B_DIM);
    k_layernorm<<<B_DIM, 256, 0, stream>>>(z3r, z3, l3g, l3b, 128, 1);
    gemm(z3, 8, bfc, hyr, B_DIM);
    k_sig<<<1, 256, 0, stream>>>(hyr, hyp, B_DIM * 4);

    // solver update
    float* Yslice = out + (size_t)k * BPN;
    k_grad_y<<<eb(BPN), 256, 0, stream>>>(Xn, Atb, yS, US, dltS, sumnb, hyp, Yslice);
    k_zero<<<eb(BPN), 256, 0, stream>>>(dltS, BPN);
    k_delta_scatter<<<dim3(B_DIM, E_DIM), 256, 0, stream>>>(yS, edge, dltS);
    k_u_update<<<eb(BPN), 256, 0, stream>>>(dltS, US, hyp);
  }
}
